// rnn_61134564491756
// MI455X (gfx1250) — matrix operand order repaired and checked by reading, not compiled, not run on hardware
//
#include <hip/hip_runtime.h>
#include <hip/hip_bf16.h>

typedef __attribute__((ext_vector_type(16))) _Float16 v16h;
typedef __attribute__((ext_vector_type(8)))  _Float16 v8h;
typedef __attribute__((ext_vector_type(8)))  float    v8f;

#define BATCH 512
#define TSTEP 256

// ---------------------------------------------------------------------------
// f32 -> f16 convert (network input x)
// ---------------------------------------------------------------------------
__global__ __launch_bounds__(256) void k_cvt_f16(const float* __restrict__ in,
                                                 _Float16* __restrict__ out, int n) {
    int i = blockIdx.x * 256 + threadIdx.x;
    if (i < n) out[i] = (_Float16)in[i];
}

// ---------------------------------------------------------------------------
// Transpose + convert: f32 [R][C] -> f16 [C][R]  (B-fragment friendly weights)
// ---------------------------------------------------------------------------
__global__ __launch_bounds__(256) void k_tcvt(const float* __restrict__ in,
                                              _Float16* __restrict__ out, int R, int C) {
    int i = blockIdx.x * 256 + threadIdx.x;
    if (i < R * C) {
        int r = i / C, c = i % C;
        out[(size_t)c * R + r] = (_Float16)in[i];
    }
}

// ---------------------------------------------------------------------------
// WMMA fragment helpers (CDNA5 ISA 7.12.2, wave32)
// A (16x32 f16): lane m=lane&15; halves -> K=kbA+[0..15] contiguous
// B (32x16 f16): lane n=lane&15; halves -> K=kbB+[0..15] contiguous
// ---------------------------------------------------------------------------
__device__ __forceinline__ v16h load_a_frag(const _Float16* p) {
    v8h lo = *(const v8h*)(p);
    v8h hi = *(const v8h*)(p + 8);
    return __builtin_shufflevector(lo, hi, 0,1,2,3,4,5,6,7,8,9,10,11,12,13,14,15);
}

// ---------------------------------------------------------------------------
// Projection GEMM, b-major M-tiles: for fixed t, 16 batch rows.
// act: f16 [B*T][K] rows=(b*T+t); wxT: f16 [N][K]; xp out: f32 [T][N][B]
// Block = 8 waves: 4 M-tiles (same t, 64 consecutive b) x 2 N-halves.
// ---------------------------------------------------------------------------
template <int K, int N>
__global__ __launch_bounds__(256) void k_proj(const _Float16* __restrict__ act,
                                              const _Float16* __restrict__ wxT,
                                              const float* __restrict__ bias,
                                              float* __restrict__ xp) {
    constexpr int NW = N / 32;
    constexpr int KK = K / 32;
    const int tid  = threadIdx.x;
    const int wave = tid >> 5, lane = tid & 31;
    const int tileid = blockIdx.x * 4 + (wave >> 1);  // 8192 tiles = T * (B/16)
    const int t      = tileid >> 5;                   // B/16 = 32 tiles per t
    const int b0     = (tileid & 31) * 16;
    const int nh     = wave & 1;
    const int lcol   = lane & 15;
    const int rowsel = (lane >> 4) * 8;
    const int kbA    = (lane >> 4) * 16;
    const int kbB    = (lane >> 4) * 16;

    v8f acc[NW];
#pragma unroll
    for (int j = 0; j < NW; ++j) {
        float bv = bias[(nh * NW + j) * 16 + lcol];
#pragma unroll
        for (int r = 0; r < 8; ++r) acc[j][r] = bv;
    }

    const _Float16* arow = act + ((size_t)(b0 + lcol) * TSTEP + t) * K;

    // two-deep pipelined fragment feed
    v16h a_cur = load_a_frag(arow + kbA);
    v16h b_cur = *(const v16h*)(wxT + (size_t)((nh * NW) * 16 + lcol) * K + kbB);
    v16h a_nxt = a_cur, b_nxt = b_cur;
#pragma unroll
    for (int kk = 0; kk < KK; ++kk) {
#pragma unroll
        for (int j = 0; j < NW; ++j) {
            if (j + 1 < NW) {
                b_nxt = *(const v16h*)(wxT + (size_t)((nh * NW + j + 1) * 16 + lcol) * K
                                       + kk * 32 + kbB);
            } else if (kk + 1 < KK) {
                a_nxt = load_a_frag(arow + (kk + 1) * 32 + kbA);
                b_nxt = *(const v16h*)(wxT + (size_t)((nh * NW) * 16 + lcol) * K
                                       + (kk + 1) * 32 + kbB);
            }
            acc[j] = __builtin_amdgcn_wmma_f32_16x16x32_f16(false, a_cur, false, b_cur,
                                                            (short)0, acc[j], false, false);
            b_cur = b_nxt;
            if (j + 1 == NW) a_cur = a_nxt;
        }
    }

    // contiguous v8f store: rows r=0..7 are consecutive b in [T][N][B]
#pragma unroll
    for (int j = 0; j < NW; ++j) {
        int colj = (nh * NW + j) * 16 + lcol;
        *(v8f*)(xp + ((size_t)t * N + colj) * BATCH + b0 + rowsel) = acc[j];
    }
}

// ---------------------------------------------------------------------------
// Recurrent scan: h_t = relu(xp_t + h_{t-1} @ Wh), 32 batch rows per WG.
// xp: f32 [T][H][B]; whT_g: f16 [H][H]; actout: f16 [B][T][H]
// 8 waves: 2 M-tiles x 4 N-quarters.
// ---------------------------------------------------------------------------
template <int H>
__global__ __launch_bounds__(256) void k_rnn_scan(const float* __restrict__ xp,
                                                  const _Float16* __restrict__ whT_g,
                                                  _Float16* __restrict__ actout) {
    constexpr int BM = 32;
    constexpr int NW = H / 64;                  // N tiles per wave
    constexpr int KK = H / 32;
    constexpr int CH = (BM * H / 8) / 256;      // copy-out chunks per thread
    extern __shared__ _Float16 smem[];
    _Float16* whl  = smem;                      // [H][H]
    _Float16* hbuf = smem + (size_t)H * H;      // [BM][H]

    const int tid  = threadIdx.x;
    const int wave = tid >> 5, lane = tid & 31;
    const int mi = wave >> 2;                   // 0..1
    const int nq = wave & 3;                    // 0..3
    const int b0 = blockIdx.x * BM;
    const int lcol   = lane & 15;
    const int rowsel = (lane >> 4) * 8;
    const int kbA    = (lane >> 4) * 16;
    const int kbB    = (lane >> 4) * 16;

    for (int i = tid; i < H * H; i += 256) whl[i] = whT_g[i];
    for (int i = tid; i < BM * H; i += 256) hbuf[i] = (_Float16)0.0f;
    __syncthreads();

    const _Float16* aprow = hbuf + (size_t)(mi * 16 + lcol) * H;

    for (int t = 0; t < TSTEP; ++t) {
        // xp for this step: contiguous f32 v8f loads, consumed only at relu time
        v8f xpv[NW];
#pragma unroll
        for (int j = 0; j < NW; ++j) {
            int colj = (nq * NW + j) * 16 + lcol;
            xpv[j] = *(const v8f*)(xp + ((size_t)t * H + colj) * BATCH + b0 + mi * 16 + rowsel);
        }

        // stream h_{t-1} to global (vectorized, overlapped with this step's MMAs)
        if (t > 0) {
#pragma unroll
            for (int c = 0; c < CH; ++c) {
                int chunk = c * 256 + tid;
                int row = chunk / (H / 8), off = (chunk % (H / 8)) * 8;
                v8h v = *(const v8h*)(hbuf + (size_t)row * H + off);
                *(v8h*)(actout + ((size_t)(b0 + row) * TSTEP + (t - 1)) * H + off) = v;
            }
        }

        // h_{t-1} @ Wh, accumulating from zero; two-deep fragment pipeline
        v8f acc[NW];
#pragma unroll
        for (int j = 0; j < NW; ++j)
#pragma unroll
            for (int r = 0; r < 8; ++r) acc[j][r] = 0.0f;

        v16h a_cur = load_a_frag(aprow + kbA);
        v16h b_cur = *(const v16h*)(whl + (size_t)((nq * NW) * 16 + lcol) * H + kbB);
        v16h a_nxt = a_cur, b_nxt = b_cur;
#pragma unroll
        for (int kk = 0; kk < KK; ++kk) {
#pragma unroll
            for (int j = 0; j < NW; ++j) {
                if (j + 1 < NW) {
                    b_nxt = *(const v16h*)(whl + (size_t)((nq * NW + j + 1) * 16 + lcol) * H
                                           + kk * 32 + kbB);
                } else if (kk + 1 < KK) {
                    a_nxt = load_a_frag(aprow + (kk + 1) * 32 + kbA);
                    b_nxt = *(const v16h*)(whl + (size_t)((nq * NW) * 16 + lcol) * H
                                           + (kk + 1) * 32 + kbB);
                }
                acc[j] = __builtin_amdgcn_wmma_f32_16x16x32_f16(false, a_cur, false, b_cur,
                                                                (short)0, acc[j], false, false);
                b_cur = b_nxt;
                if (j + 1 == NW) a_cur = a_nxt;
            }
        }
        __syncthreads();                        // all reads of h_{t-1} complete

        // h_t = relu(mma + xp): write back to LDS only (global copy next step)
#pragma unroll
        for (int j = 0; j < NW; ++j) {
            int colj = (nq * NW + j) * 16 + lcol;
#pragma unroll
            for (int r = 0; r < 8; ++r) {
                float v = fmaxf(acc[j][r] + xpv[j][r], 0.0f);
                hbuf[(size_t)(mi * 16 + r + rowsel) * H + colj] = (_Float16)v;
            }
        }
        __syncthreads();                        // h_t visible before next step
    }

    // epilogue: stream h_{T-1}
#pragma unroll
    for (int c = 0; c < CH; ++c) {
        int chunk = c * 256 + tid;
        int row = chunk / (H / 8), off = (chunk % (H / 8)) * 8;
        v8h v = *(const v8h*)(hbuf + (size_t)row * H + off);
        *(v8h*)(actout + ((size_t)(b0 + row) * TSTEP + (TSTEP - 1)) * H + off) = v;
    }
}

// ---------------------------------------------------------------------------
// Dense head: out[b] = flat(h4)[b,:] @ Wd + bd  (dot length T*64 = 16384)
// ---------------------------------------------------------------------------
__global__ __launch_bounds__(256) void k_dense(const _Float16* __restrict__ h,
                                               const float* __restrict__ Wd,
                                               const float* __restrict__ bd,
                                               float* __restrict__ out) {
    __shared__ float red[256];
    const int b = blockIdx.x, tid = threadIdx.x;
    const _Float16* hp = h + (size_t)b * (TSTEP * 64);
    float s = 0.0f;
    for (int i = tid; i < TSTEP * 64; i += 256) s += (float)hp[i] * Wd[i];
    red[tid] = s;
    __syncthreads();
    for (int off = 128; off; off >>= 1) {
        if (tid < off) red[tid] += red[tid + off];
        __syncthreads();
    }
    if (tid == 0) out[b] = red[0] + bd[0];
}

// ---------------------------------------------------------------------------
// Host driver
// ---------------------------------------------------------------------------
extern "C" void kernel_launch(void* const* d_in, const int* in_sizes, int n_in,
                              void* d_out, int out_size, void* d_ws, size_t ws_size,
                              hipStream_t stream) {
    (void)in_sizes; (void)n_in; (void)out_size; (void)ws_size;

    const float* x  = (const float*)d_in[0];
    const float* Wx[4]   = {(const float*)d_in[1], (const float*)d_in[4],
                            (const float*)d_in[7], (const float*)d_in[10]};
    const float* Wh[4]   = {(const float*)d_in[2], (const float*)d_in[5],
                            (const float*)d_in[8], (const float*)d_in[11]};
    const float* bias[4] = {(const float*)d_in[3], (const float*)d_in[6],
                            (const float*)d_in[9], (const float*)d_in[12]};
    const float* Wd = (const float*)d_in[13];
    const float* bd = (const float*)d_in[14];
    float* out = (float*)d_out;

    // workspace: actA/actB f16 (64 MB each), xp f32 (128 MB), weight staging
    const size_t ACT_BYTES = (size_t)BATCH * TSTEP * 256 * sizeof(_Float16);
    char* ws = (char*)d_ws;
    _Float16* actA = (_Float16*)(ws);
    _Float16* actB = (_Float16*)(ws + ACT_BYTES);
    float*    xp   = (float*)(ws + 2 * ACT_BYTES);
    _Float16* wxT  = (_Float16*)(ws + 2 * ACT_BYTES + (size_t)TSTEP * BATCH * 256 * 4);
    _Float16* whT  = wxT + 256 * 256;

    (void)hipFuncSetAttribute((const void*)k_rnn_scan<256>,
                              hipFuncAttributeMaxDynamicSharedMemorySize,
                              (256 * 256 + 32 * 256) * (int)sizeof(_Float16));
    (void)hipFuncSetAttribute((const void*)k_rnn_scan<128>,
                              hipFuncAttributeMaxDynamicSharedMemorySize,
                              (128 * 128 + 32 * 128) * (int)sizeof(_Float16));
    (void)hipFuncSetAttribute((const void*)k_rnn_scan<64>,
                              hipFuncAttributeMaxDynamicSharedMemorySize,
                              (64 * 64 + 32 * 64) * (int)sizeof(_Float16));

    const int PROJ_GRID = (TSTEP * (BATCH / 16)) / 4;   // 2048
    const int SCAN_GRID = BATCH / 32;                   // 16

    { // x -> f16
        int n = BATCH * TSTEP * 64;
        k_cvt_f16<<<(n + 255) / 256, 256, 0, stream>>>(x, actA, n);
    }

    // ---- layer 1: 64 -> 128 ----
    k_tcvt<<<(64 * 128 + 255) / 256, 256, 0, stream>>>(Wx[0], wxT, 64, 128);
    k_tcvt<<<(128 * 128 + 255) / 256, 256, 0, stream>>>(Wh[0], whT, 128, 128);
    k_proj<64, 128><<<PROJ_GRID, 256, 0, stream>>>(actA, wxT, bias[0], xp);
    k_rnn_scan<128><<<SCAN_GRID, 256,
                      (128 * 128 + 32 * 128) * sizeof(_Float16), stream>>>(xp, whT, actB);

    // ---- layer 2: 128 -> 256 ----
    k_tcvt<<<(128 * 256 + 255) / 256, 256, 0, stream>>>(Wx[1], wxT, 128, 256);
    k_tcvt<<<(256 * 256 + 255) / 256, 256, 0, stream>>>(Wh[1], whT, 256, 256);
    k_proj<128, 256><<<PROJ_GRID, 256, 0, stream>>>(actB, wxT, bias[1], xp);
    k_rnn_scan<256><<<SCAN_GRID, 256,
                      (256 * 256 + 32 * 256) * sizeof(_Float16), stream>>>(xp, whT, actA);

    // ---- layer 3: 256 -> 128 ----
    k_tcvt<<<(256 * 128 + 255) / 256, 256, 0, stream>>>(Wx[2], wxT, 256, 128);
    k_tcvt<<<(128 * 128 + 255) / 256, 256, 0, stream>>>(Wh[2], whT, 128, 128);
    k_proj<256, 128><<<PROJ_GRID, 256, 0, stream>>>(actA, wxT, bias[2], xp);
    k_rnn_scan<128><<<SCAN_GRID, 256,
                      (128 * 128 + 32 * 128) * sizeof(_Float16), stream>>>(xp, whT, actB);

    // ---- layer 4: 128 -> 64 ----
    k_tcvt<<<(128 * 64 + 255) / 256, 256, 0, stream>>>(Wx[3], wxT, 128, 64);
    k_tcvt<<<(64 * 64 + 255) / 256, 256, 0, stream>>>(Wh[3], whT, 64, 64);
    k_proj<128, 64><<<PROJ_GRID, 256, 0, stream>>>(actB, wxT, bias[3], xp);
    k_rnn_scan<64><<<SCAN_GRID, 256,
                     (64 * 64 + 32 * 64) * sizeof(_Float16), stream>>>(xp, whT, actA);

    // ---- dense head ----
    k_dense<<<BATCH, 256, 0, stream>>>(actA, Wd, bd, out);
}
